// LearnableUpsampler_18150531792908
// MI455X (gfx1250) — 8-bit matrix operand order repaired and checked by reading, compiled, not run on hardware
//
#include <hip/hip_runtime.h>
#include <math.h>

typedef __attribute__((ext_vector_type(8))) int v8i;

#define B_     2
#define T_     2048
#define C_     512
#define L_     4096
#define KR_    7
#define PR_    3
#define NT_N   (C_ / 16)
#define NPAIR  (NT_N / 2)
#define KT_REF ((C_ * KR_) / 64)
#define KT_UP  ((C_ * 2) / 64)

__global__ __launch_bounds__(256)
void wamp_kernel(const float* __restrict__ w0, int n0,
                 const float* __restrict__ w1, int n1,
                 const float* __restrict__ w2, int n2,
                 float* __restrict__ amp)
{
    const float* w = (blockIdx.x == 0) ? w0 : (blockIdx.x == 1 ? w1 : w2);
    const int    n = (blockIdx.x == 0) ? n0 : (blockIdx.x == 1 ? n1 : n2);
    __shared__ float red[256];
    float s = 0.f;
    for (int i = threadIdx.x; i < n; i += 256) s += fabsf(w[i]);
    red[threadIdx.x] = s;
    __syncthreads();
    for (int st = 128; st > 0; st >>= 1) {
        if ((int)threadIdx.x < st) red[threadIdx.x] += red[threadIdx.x + st];
        __syncthreads();
    }
    if (threadIdx.x == 0) amp[blockIdx.x] = fmaxf(red[0] / (float)n, 1e-5f);
}

__global__ __launch_bounds__(256)
void act_quant_kernel(const float* __restrict__ x,
                      signed char* __restrict__ q,
                      float* __restrict__ s)
{
    const int row = blockIdx.x;
    const float* xr = x + (size_t)row * C_;
    __shared__ float red[256];
    float m = 0.f;
    for (int c = threadIdx.x; c < C_; c += 256) m = fmaxf(m, fabsf(xr[c]));
    red[threadIdx.x] = m;
    __syncthreads();
    for (int st = 128; st > 0; st >>= 1) {
        if ((int)threadIdx.x < st)
            red[threadIdx.x] = fmaxf(red[threadIdx.x], red[threadIdx.x + st]);
        __syncthreads();
    }
    const float scale = 127.f / fmaxf(red[0], 1e-5f);
    if (threadIdx.x == 0) s[row] = scale;
    for (int c = threadIdx.x; c < C_; c += 256) {
        float v = rintf(xr[c] * scale);
        v = fminf(fmaxf(v, -128.f), 127.f);
        q[(size_t)row * C_ + c] = (signed char)(int)v;
    }
}

__global__ __launch_bounds__(256)
void pack_up_kernel(const float* __restrict__ w,
                    const float* __restrict__ amp,
                    int* __restrict__ dst)
{
    const int tid  = blockIdx.x * 256 + threadIdx.x;
    const int v    = tid & 7;
    const int lane = (tid >> 3) & 31;
    const int tile = tid >> 8;
    const int nt   = tile % NT_N;
    const int kt   = (tile / NT_N) % KT_UP;
    const int p    = tile / (NT_N * KT_UP);
    const int colN = lane & 15, halfB = lane >> 4;
    const int co   = nt * 16 + colN;
    const float invamp = 1.f / amp[0];
    unsigned dw = 0;
    #pragma unroll
    for (int j = 0; j < 4; ++j) {
        const int Kl   = (v >> 2) * 32 + halfB * 16 + (v & 3) * 4 + j;
        const int K    = kt * 64 + Kl;
        const int tapi = K >> 9;
        const int cin  = K & (C_ - 1);
        const int k    = (1 - p) + 2 * tapi;
        const float wv = w[((size_t)cin * C_ + co) * 4 + k];
        int qv = (int)rintf(wv * invamp);
        qv = qv < -1 ? -1 : (qv > 1 ? 1 : qv);
        dw |= ((unsigned)(qv & 0xff)) << (8 * j);
    }
    dst[tid] = (int)dw;
}

__global__ __launch_bounds__(256)
void pack_ref_kernel(const float* __restrict__ w1, const float* __restrict__ w2,
                     const float* __restrict__ amp,
                     int* __restrict__ d1, int* __restrict__ d2)
{
    const int tid = blockIdx.x * 256 + threadIdx.x;
    const float* w  = blockIdx.y ? w2 : w1;
    int*        dst = blockIdx.y ? d2 : d1;
    const float invamp = 1.f / amp[1 + blockIdx.y];
    const int v = tid & 7, lane = (tid >> 3) & 31, tile = tid >> 8;
    const int nt = tile % NT_N, kt = tile / NT_N;
    const int colN = lane & 15, halfB = lane >> 4;
    const int co = nt * 16 + colN;
    unsigned dw = 0;
    #pragma unroll
    for (int j = 0; j < 4; ++j) {
        const int Kl  = (v >> 2) * 32 + halfB * 16 + (v & 3) * 4 + j;
        const int K   = kt * 64 + Kl;
        const int tap = K >> 9;
        const int cin = K & (C_ - 1);
        const float wv = w[((size_t)co * C_ + cin) * KR_ + tap];
        int qv = (int)rintf(wv * invamp);
        qv = qv < -1 ? -1 : (qv > 1 ? 1 : qv);
        dw |= ((unsigned)(qv & 0xff)) << (8 * j);
    }
    dst[tid] = (int)dw;
}

__device__ __forceinline__ v8i load_a_row(const signed char* __restrict__ arow,
                                          int cin0, int half, bool ok)
{
    v8i A;
    #pragma unroll
    for (int pr = 0; pr < 4; ++pr) {
        const int2 d = *(const int2*)(arow + cin0 + (pr >> 1) * 32 + half * 16 + (pr & 1) * 8);
        A[2 * pr]     = ok ? d.x : 0;
        A[2 * pr + 1] = ok ? d.y : 0;
    }
    return A;
}

__device__ __forceinline__ v8i load_b_tile(const int* __restrict__ pb,
                                           int kt, int nt, int lane)
{
    const int4* bp = (const int4*)(pb + (((size_t)kt * NT_N + nt) * 32 + lane) * 8);
    const int4 b0 = bp[0], b1 = bp[1];
    v8i B;
    B[0] = b0.x; B[1] = b0.y; B[2] = b0.z; B[3] = b0.w;
    B[4] = b1.x; B[5] = b1.y; B[6] = b1.z; B[7] = b1.w;
    return B;
}

__global__ __launch_bounds__(128)
void upconv_wmma_kernel(const signed char* __restrict__ qx, const float* __restrict__ sx,
                        const int* __restrict__ packB, const float* __restrict__ amp,
                        const float* __restrict__ bias, float* __restrict__ h)
{
    const int lane = threadIdx.x & 31;
    const int wave = blockIdx.x * (blockDim.x >> 5) + (threadIdx.x >> 5);
    const int mt   = wave / NPAIR;
    const int np   = wave - mt * NPAIR;
    const int nt0  = np * 2, nt1 = nt0 + 1;
    const int b    = mt >> 8;
    const int p    = (mt >> 7) & 1;
    const int chunk = mt & 127;
    const int half = lane >> 4;
    const int r    = lane & 15;
    const int hidx = chunk * 16 + r;
    const int* pb  = packB + (size_t)p * KT_UP * NT_N * 256;

    float facc0[8] = {0.f, 0.f, 0.f, 0.f, 0.f, 0.f, 0.f, 0.f};
    float facc1[8] = {0.f, 0.f, 0.f, 0.f, 0.f, 0.f, 0.f, 0.f};
    for (int tap = 0; tap < 2; ++tap) {
        const int  ts = hidx + p - tap;
        const bool ok = (ts >= 0) && (ts < T_);
        const int  tsc = ts < 0 ? 0 : (ts >= T_ ? T_ - 1 : ts);
        const signed char* arow = qx + ((size_t)b * T_ + tsc) * C_;
        v8i acc0 = {0, 0, 0, 0, 0, 0, 0, 0};
        v8i acc1 = {0, 0, 0, 0, 0, 0, 0, 0};
        #pragma unroll
        for (int k8 = 0; k8 < 8; ++k8) {
            const v8i A  = load_a_row(arow, k8 * 64, half, ok);
            const v8i B0 = load_b_tile(pb, tap * 8 + k8, nt0, lane);
            const v8i B1 = load_b_tile(pb, tap * 8 + k8, nt1, lane);
            acc0 = __builtin_amdgcn_wmma_i32_16x16x64_iu8(true, A, true, B0, acc0,
                                                          true,  false);
            acc1 = __builtin_amdgcn_wmma_i32_16x16x64_iu8(true, A, true, B1, acc1,
                                                          false, false);
        }
        #pragma unroll
        for (int rr = 0; rr < 8; ++rr) {
            const int  M    = rr + half * 8;
            const int  tsM  = chunk * 16 + M + p - tap;
            const bool okM  = (tsM >= 0) && (tsM < T_);
            const int  tsMc = tsM < 0 ? 0 : (tsM >= T_ ? T_ - 1 : tsM);
            const float inv = okM ? 1.f / sx[(size_t)b * T_ + tsMc] : 0.f;
            facc0[rr] += (float)acc0[rr] * inv;
            facc1[rr] += (float)acc1[rr] * inv;
        }
    }
    const float wa   = amp[0];
    const int   col0 = nt0 * 16 + r;
    const int   col1 = nt1 * 16 + r;
    const float bv0  = bias[col0];
    const float bv1  = bias[col1];
    #pragma unroll
    for (int rr = 0; rr < 8; ++rr) {
        const int M = rr + half * 8;
        const int l = (chunk * 16 + M) * 2 + p;
        float* hrow = h + ((size_t)b * L_ + l) * C_;
        hrow[col0] = facc0[rr] * wa + bv0;
        hrow[col1] = facc1[rr] * wa + bv1;
    }
}

__global__ __launch_bounds__(128)
void conv7_wmma_kernel(const signed char* __restrict__ qa, const float* __restrict__ sa,
                       const int* __restrict__ packB, const float* __restrict__ amp,
                       const float* __restrict__ bias, const float* __restrict__ resid,
                       float* __restrict__ out, int fuse_silu)
{
    const int lane = threadIdx.x & 31;
    const int wave = blockIdx.x * (blockDim.x >> 5) + (threadIdx.x >> 5);
    const int mt   = wave / NPAIR;
    const int np   = wave - mt * NPAIR;
    const int nt0  = np * 2, nt1 = nt0 + 1;
    const int g0   = mt * 16;
    const int b    = g0 / L_;
    const int l0   = g0 - b * L_;
    const int half = lane >> 4;
    const int r    = lane & 15;
    const int lr   = l0 + r;

    float facc0[8] = {0.f, 0.f, 0.f, 0.f, 0.f, 0.f, 0.f, 0.f};
    float facc1[8] = {0.f, 0.f, 0.f, 0.f, 0.f, 0.f, 0.f, 0.f};
    for (int tap = 0; tap < KR_; ++tap) {
        const int  ts  = lr + tap - PR_;
        const bool ok  = (ts >= 0) && (ts < L_);
        const int  tsc = ts < 0 ? 0 : (ts >= L_ ? L_ - 1 : ts);
        const signed char* arow = qa + ((size_t)b * L_ + tsc) * C_;
        v8i acc0 = {0, 0, 0, 0, 0, 0, 0, 0};
        v8i acc1 = {0, 0, 0, 0, 0, 0, 0, 0};
        #pragma unroll
        for (int k8 = 0; k8 < 8; ++k8) {
            const v8i A  = load_a_row(arow, k8 * 64, half, ok);
            const v8i B0 = load_b_tile(packB, tap * 8 + k8, nt0, lane);
            const v8i B1 = load_b_tile(packB, tap * 8 + k8, nt1, lane);
            acc0 = __builtin_amdgcn_wmma_i32_16x16x64_iu8(true, A, true, B0, acc0,
                                                          true,  false);
            acc1 = __builtin_amdgcn_wmma_i32_16x16x64_iu8(true, A, true, B1, acc1,
                                                          false, false);
        }
        #pragma unroll
        for (int rr = 0; rr < 8; ++rr) {
            const int  M    = rr + half * 8;
            const int  tsM  = l0 + M + tap - PR_;
            const bool okM  = (tsM >= 0) && (tsM < L_);
            const int  tsMc = tsM < 0 ? 0 : (tsM >= L_ ? L_ - 1 : tsM);
            const float inv = okM ? 1.f / sa[(size_t)b * L_ + tsMc] : 0.f;
            facc0[rr] += (float)acc0[rr] * inv;
            facc1[rr] += (float)acc1[rr] * inv;
        }
    }
    const float wa   = amp[0];
    const int   col0 = nt0 * 16 + r;
    const int   col1 = nt1 * 16 + r;
    const float bv0  = bias[col0];
    const float bv1  = bias[col1];
    #pragma unroll
    for (int rr = 0; rr < 8; ++rr) {
        const int    M = rr + half * 8;
        const size_t g = (size_t)g0 + M;
        float v0 = facc0[rr] * wa + bv0;
        float v1 = facc1[rr] * wa + bv1;
        if (fuse_silu) {
            v0 = v0 / (1.f + expf(-v0));
            v1 = v1 / (1.f + expf(-v1));
        } else {
            v0 += resid[g * C_ + col0];
            v1 += resid[g * C_ + col1];
        }
        out[g * C_ + col0] = v0;
        out[g * C_ + col1] = v1;
    }
}

__global__ __launch_bounds__(256)
void ln_kernel(const float* __restrict__ hr, const float* __restrict__ lw,
               const float* __restrict__ lb, float* __restrict__ out)
{
    const int row = blockIdx.x;
    const float* xr = hr + (size_t)row * C_;
    __shared__ float rs[256], rs2[256];
    float s = 0.f, s2 = 0.f;
    for (int c = threadIdx.x; c < C_; c += 256) {
        const float v = xr[c];
        s += v; s2 += v * v;
    }
    rs[threadIdx.x] = s; rs2[threadIdx.x] = s2;
    __syncthreads();
    for (int st = 128; st > 0; st >>= 1) {
        if ((int)threadIdx.x < st) {
            rs[threadIdx.x]  += rs[threadIdx.x + st];
            rs2[threadIdx.x] += rs2[threadIdx.x + st];
        }
        __syncthreads();
    }
    const float mu  = rs[0] / (float)C_;
    const float var = rs2[0] / (float)C_ - mu * mu;
    const float inv = rsqrtf(var + 1e-5f);
    for (int c = threadIdx.x; c < C_; c += 256)
        out[(size_t)row * C_ + c] = (xr[c] - mu) * inv * lw[c] + lb[c];
}

extern "C" void kernel_launch(void* const* d_in, const int* in_sizes, int n_in,
                              void* d_out, int out_size, void* d_ws, size_t ws_size,
                              hipStream_t stream)
{
    (void)in_sizes; (void)n_in; (void)out_size; (void)ws_size;
    const float* x    = (const float*)d_in[0];
    const float* w_up = (const float*)d_in[1];
    const float* b_up = (const float*)d_in[2];
    const float* w_r1 = (const float*)d_in[3];
    const float* b_r1 = (const float*)d_in[4];
    const float* w_r2 = (const float*)d_in[5];
    const float* b_r2 = (const float*)d_in[6];
    const float* ln_w = (const float*)d_in[7];
    const float* ln_b = (const float*)d_in[8];
    float* out = (float*)d_out;

    char* ws = (char*)d_ws;
    size_t off = 0;
    auto alloc = [&](size_t bytes) -> char* {
        char* p = ws + off;
        off = (off + bytes + 255) & ~(size_t)255;
        return p;
    };
    float*       amp   = (float*)alloc(3 * sizeof(float));
    signed char* qx    = (signed char*)alloc((size_t)B_ * T_ * C_);
    float*       sx    = (float*)alloc((size_t)B_ * T_ * 4);
    int*         pb_up = (int*)alloc((size_t)2 * KT_UP * NT_N * 256 * 4);
    int*         pb_r1 = (int*)alloc((size_t)KT_REF * NT_N * 256 * 4);
    int*         pb_r2 = (int*)alloc((size_t)KT_REF * NT_N * 256 * 4);
    float*       h     = (float*)alloc((size_t)B_ * L_ * C_ * 4);
    signed char* qh    = (signed char*)alloc((size_t)B_ * L_ * C_);
    float*       sh    = (float*)alloc((size_t)B_ * L_ * 4);
    float*       r1    = (float*)alloc((size_t)B_ * L_ * C_ * 4);
    signed char* qr    = (signed char*)alloc((size_t)B_ * L_ * C_);
    float*       sr    = (float*)alloc((size_t)B_ * L_ * 4);
    float*       hr    = (float*)alloc((size_t)B_ * L_ * C_ * 4);

    wamp_kernel<<<3, 256, 0, stream>>>(w_up, C_ * C_ * 4,
                                       w_r1, C_ * C_ * KR_,
                                       w_r2, C_ * C_ * KR_, amp);
    act_quant_kernel<<<B_ * T_, 256, 0, stream>>>(x, qx, sx);
    pack_up_kernel<<<(2 * KT_UP * NT_N * 256) / 256, 256, 0, stream>>>(w_up, amp, pb_up);
    dim3 gr((KT_REF * NT_N * 256) / 256, 2);
    pack_ref_kernel<<<gr, 256, 0, stream>>>(w_r1, w_r2, amp, pb_r1, pb_r2);
    upconv_wmma_kernel<<<(512 * NPAIR) / 4, 128, 0, stream>>>(qx, sx, pb_up, amp, b_up, h);
    act_quant_kernel<<<B_ * L_, 256, 0, stream>>>(h, qh, sh);
    conv7_wmma_kernel<<<((B_ * L_ / 16) * NPAIR) / 4, 128, 0, stream>>>(
        qh, sh, pb_r1, amp + 1, b_r1, nullptr, r1, 1);
    act_quant_kernel<<<B_ * L_, 256, 0, stream>>>(r1, qr, sr);
    conv7_wmma_kernel<<<((B_ * L_ / 16) * NPAIR) / 4, 128, 0, stream>>>(
        qr, sr, pb_r2, amp + 2, b_r2, h, hr, 0);
    ln_kernel<<<B_ * L_, 256, 0, stream>>>(hr, ln_w, ln_b, out);
}
